// AttentionDeformModule_26079041421783
// MI455X (gfx1250) — matrix operand order repaired and checked by reading, not compiled, not run on hardware
//
#include <hip/hip_runtime.h>
#include <hip/hip_bf16.h>

// ---------------- problem constants (from reference) ----------------
#define BATCH 2
#define CDIM 256
#define PNUM 2
#define KBINS 5
#define HDIM 100
#define WDIM 152
#define HWN (HDIM * WDIM)        // 15200
#define DDIM (PNUM * CDIM)       // 512
#define NTILES (HWN / 16)        // 950 (exact)
#define GN_GROUPS 32
#define GN_EPS 1e-5f
#define EPS_DIV 1e-6f

// ---------------- CDNA5 WMMA types ----------------
typedef __attribute__((ext_vector_type(16))) __bf16 v16bf;
typedef __attribute__((ext_vector_type(8)))  float  v8f;

union FragU {
  v16bf v;
  unsigned u[8];
};

__device__ __forceinline__ __bf16 f2bf(float f) { return (__bf16)f; }
__device__ __forceinline__ float bf2f(__bf16 h) { return (float)h; }

// A: row-major M x K (lda = K). CDNA5 16-bit A 16x32 fragment layout:
// lanes 0-15 row M=lane hold K = {2j,2j+1} (V0-3) and {16+2j} (V4-7);
// lanes 16-31 same rows, K offset +8.
__device__ __forceinline__ void load_a_frag(const __bf16* A, int lda, int m0,
                                            int k0, int r, int half, FragU& a) {
  const __bf16* row = A + (size_t)(m0 + r) * lda + k0 + 8 * half;
#pragma unroll
  for (int j = 0; j < 4; ++j)
    a.u[j] = *(const unsigned*)(row + 2 * j);
#pragma unroll
  for (int j = 0; j < 4; ++j)
    a.u[4 + j] = *(const unsigned*)(row + 16 + 2 * j);
}

// B stored N-major (Bt is N x K, ldb = K). CDNA5 16-bit B 32x16 layout:
// lane = column N; lanes 0-15 hold K=0..7 (V0-3) and 16..23 (V4-7), lanes 16-31 K offset +8.
__device__ __forceinline__ void load_b_frag(const __bf16* Bt, int ldb, int n0,
                                            int k0, int r, int half, FragU& b) {
  const __bf16* row = Bt + (size_t)(n0 + r) * ldb + k0 + 8 * half;
#pragma unroll
  for (int j = 0; j < 8; ++j)
    b.u[j] = *(const unsigned*)(row + (j < 4 ? 2 * j : 16 + 2 * (j - 4)));
}

// ---------------- conversion / init kernels ----------------
__global__ void __launch_bounds__(256) cvt_kernel(const float* __restrict__ s,
                                                  __bf16* __restrict__ d, int n) {
  int i = blockIdx.x * blockDim.x + threadIdx.x;
  if (i < n) d[i] = f2bf(s[i]);
}

// x (B,C,HW) f32 -> xt (B,HW,C) bf16  (writes coalesced over channel)
__global__ void __launch_bounds__(256) cvt_x_kernel(const float* __restrict__ x,
                                                    __bf16* __restrict__ xt) {
  int i = blockIdx.x * blockDim.x + threadIdx.x;
  if (i >= BATCH * HWN * CDIM) return;
  int c = i % CDIM;
  int rem = i / CDIM;
  int n = rem % HWN;
  int b = rem / HWN;
  xt[i] = f2bf(x[((size_t)b * CDIM + c) * HWN + n]);
}

__global__ void __launch_bounds__(256) zero_kernel(float* __restrict__ p, int n) {
  int i = blockIdx.x * blockDim.x + threadIdx.x;
  if (i < n) p[i] = 0.f;
}

// ---------------- GEMM 1: ef = ef_w (512x256) x X (256xHW) + ef_b ----------------
// Output N-major bf16: eft[b][n][d]
__global__ void __launch_bounds__(256) gemm_ef_kernel(const __bf16* __restrict__ W,
                                                      const __bf16* __restrict__ Xt,
                                                      const float* __restrict__ bias,
                                                      __bf16* __restrict__ Yt) {
  int wave = (blockIdx.x * blockDim.x + threadIdx.x) >> 5;
  int lane = threadIdx.x & 31;
  int r = lane & 15, half = lane >> 4;
  int nt = wave % NTILES;
  int rem = wave / NTILES;
  int mt = rem % (DDIM / 16);
  int b = rem / (DDIM / 16);
  int m0 = mt * 16, n0 = nt * 16;
  const __bf16* Bt = Xt + (size_t)b * HWN * CDIM;
  FragU a, bf;
  v8f acc = {};
#pragma unroll
  for (int k0 = 0; k0 < CDIM; k0 += 32) {
    if (k0 + 32 < CDIM)
      __builtin_prefetch(Bt + (size_t)(n0 + r) * CDIM + k0 + 32, 0, 1);
    load_a_frag(W, CDIM, m0, k0, r, half, a);
    load_b_frag(Bt, CDIM, n0, k0, r, half, bf);
    acc = __builtin_amdgcn_wmma_f32_16x16x32_bf16(false, a.v, false, bf.v,
                                                  (short)0, acc, false, false);
  }
  int mbase = m0 + 8 * half;  // C/D layout: VGPR v -> row mbase+v, col n0+r
  union { __bf16 e[8]; uint4 q; } o;
#pragma unroll
  for (int v = 0; v < 8; ++v)
    o.e[v] = f2bf(acc[v] + bias[mbase + v]);
  *(uint4*)(Yt + (size_t)b * HWN * DDIM + (size_t)(n0 + r) * DDIM + mbase) = o.q;
}

// ---------------- GEMM 2 (fused): heat accumulation ----------------
// h = relu(hm1_w[p] x efp + hm1_b[p]); heat_acc[b,p,n] += sum_d h[d,n]*hm2_w[p,d]
__global__ void __launch_bounds__(256) gemm_heat_kernel(const __bf16* __restrict__ Wh,
                                                        const __bf16* __restrict__ EFt,
                                                        const float* __restrict__ b1,
                                                        const float* __restrict__ w2,
                                                        float* __restrict__ heat) {
  int wave = (blockIdx.x * blockDim.x + threadIdx.x) >> 5;
  int lane = threadIdx.x & 31;
  int r = lane & 15, half = lane >> 4;
  int nt = wave % NTILES;
  int rem = wave / NTILES;
  int mt = rem % (CDIM / 16);
  rem /= (CDIM / 16);
  int p = rem % PNUM;
  int b = rem / PNUM;
  int m0 = mt * 16, n0 = nt * 16;
  const __bf16* A = Wh + (size_t)p * CDIM * CDIM;
  const __bf16* Bt = EFt + (size_t)b * HWN * DDIM + p * CDIM;
  FragU a, bf;
  v8f acc = {};
#pragma unroll
  for (int k0 = 0; k0 < CDIM; k0 += 32) {
    if (k0 + 32 < CDIM)
      __builtin_prefetch(Bt + (size_t)(n0 + r) * DDIM + k0 + 32, 0, 1);
    load_a_frag(A, CDIM, m0, k0, r, half, a);
    load_b_frag(Bt, DDIM, n0, k0, r, half, bf);
    acc = __builtin_amdgcn_wmma_f32_16x16x32_bf16(false, a.v, false, bf.v,
                                                  (short)0, acc, false, false);
  }
  int mbase = m0 + 8 * half;
  float s = 0.f;
#pragma unroll
  for (int v = 0; v < 8; ++v) {
    int m = mbase + v;
    float hv = fmaxf(acc[v] + b1[p * CDIM + m], 0.f);
    s += hv * w2[p * CDIM + m];
  }
  atomicAdd(&heat[((size_t)b * PNUM + p) * HWN + n0 + r], s);
}

__global__ void __launch_bounds__(256) heat_exp_kernel(float* __restrict__ heat,
                                                       const float* __restrict__ b2) {
  int i = blockIdx.x * blockDim.x + threadIdx.x;
  if (i < BATCH * PNUM * HWN) {
    int p = (i / HWN) % PNUM;
    heat[i] = expf(heat[i] + b2[p]);
  }
}

// ---------------- deformable sampling (gather, channel-contiguous) ----------------
// One block per (b,p,pixel): 20 (idx, w*heat) pairs precomputed in LDS,
// 256 threads = 256 channels, coalesced bf16 gathers from N-major eft.
__global__ void __launch_bounds__(256) sample_kernel(const __bf16* __restrict__ EFt,
                                                     const float* __restrict__ heat,
                                                     const float* __restrict__ offs,
                                                     __bf16* __restrict__ St) {
  int pid = blockIdx.x;
  int n = pid % HWN;
  int p = (pid / HWN) % PNUM;
  int b = pid / (HWN * PNUM);
  __shared__ float wts[KBINS * 4];
  __shared__ int sidx[KBINS * 4];
  int tid = threadIdx.x;
  if (tid < KBINS * 4) {
    int k = tid >> 2;
    int dy = (tid >> 1) & 1;
    int dx = tid & 1;
    int hh = n / WDIM, ww = n % WDIM;
    size_t obase =
        ((size_t)b * (PNUM * KBINS * 2) + (size_t)(p * KBINS + k) * 2) * HWN + n;
    float ys = (float)hh + offs[obase];
    float xs = (float)ww + offs[obase + HWN];
    float yi = floorf(ys) + (float)dy;
    float xi = floorf(xs) + (float)dx;
    float w = (1.f - fabsf(ys - yi)) * (1.f - fabsf(xs - xi));
    bool valid = (yi >= 0.f) && (yi <= (float)(HDIM - 1)) && (xi >= 0.f) &&
                 (xi <= (float)(WDIM - 1));
    int yc = min(max((int)yi, 0), HDIM - 1);
    int xc = min(max((int)xi, 0), WDIM - 1);
    int idx = yc * WDIM + xc;
    float hv = heat[((size_t)b * PNUM + p) * HWN + idx];
    wts[tid] = valid ? w * hv : 0.f;
    sidx[tid] = idx;
  }
  __syncthreads();
  int c = tid;
  const __bf16* base = EFt + (size_t)b * HWN * DDIM + p * CDIM + c;
  float num = 0.f, den = 0.f;
#pragma unroll
  for (int t = 0; t < KBINS * 4; ++t) {
    float w = wts[t];
    den += w;
    num += w * bf2f(base[(size_t)sidx[t] * DDIM]);
  }
  St[((size_t)b * HWN + n) * DDIM + p * CDIM + c] = f2bf(num / (den + EPS_DIV));
}

// ---------------- GEMM 3: merged = merge_w (256x512) x outs (512xHW) + merge_b ----
// Output D-major fp32 for GroupNorm: merged[b][d][n]
__global__ void __launch_bounds__(256) gemm_merge_kernel(const __bf16* __restrict__ Wm,
                                                         const __bf16* __restrict__ St,
                                                         const float* __restrict__ bias,
                                                         float* __restrict__ Y) {
  int wave = (blockIdx.x * blockDim.x + threadIdx.x) >> 5;
  int lane = threadIdx.x & 31;
  int r = lane & 15, half = lane >> 4;
  int nt = wave % NTILES;
  int rem = wave / NTILES;
  int mt = rem % (CDIM / 16);
  int b = rem / (CDIM / 16);
  int m0 = mt * 16, n0 = nt * 16;
  const __bf16* Bt = St + (size_t)b * HWN * DDIM;
  FragU a, bf;
  v8f acc = {};
#pragma unroll
  for (int k0 = 0; k0 < DDIM; k0 += 32) {
    if (k0 + 32 < DDIM)
      __builtin_prefetch(Bt + (size_t)(n0 + r) * DDIM + k0 + 32, 0, 1);
    load_a_frag(Wm, DDIM, m0, k0, r, half, a);
    load_b_frag(Bt, DDIM, n0, k0, r, half, bf);
    acc = __builtin_amdgcn_wmma_f32_16x16x32_bf16(false, a.v, false, bf.v,
                                                  (short)0, acc, false, false);
  }
  int mbase = m0 + 8 * half;
#pragma unroll
  for (int v = 0; v < 8; ++v) {
    int m = mbase + v;
    Y[((size_t)b * CDIM + m) * HWN + n0 + r] = acc[v] + bias[m];
  }
}

// ---------------- GroupNorm(32) + ReLU ----------------
__global__ void __launch_bounds__(256) gn_reduce_kernel(const float* __restrict__ merged,
                                                        float* __restrict__ stats) {
  int bg = blockIdx.x;
  int b = bg / GN_GROUPS;
  int g = bg % GN_GROUPS;
  const float* p = merged + ((size_t)b * CDIM + g * (CDIM / GN_GROUPS)) * HWN;
  const int n = (CDIM / GN_GROUPS) * HWN;
  float s = 0.f, ss = 0.f;
  for (int i = threadIdx.x; i < n; i += blockDim.x) {
    float v = p[i];
    s += v;
    ss += v * v;
  }
  __shared__ float sh[256], sh2[256];
  sh[threadIdx.x] = s;
  sh2[threadIdx.x] = ss;
  __syncthreads();
  for (int st = 128; st > 0; st >>= 1) {
    if (threadIdx.x < st) {
      sh[threadIdx.x] += sh[threadIdx.x + st];
      sh2[threadIdx.x] += sh2[threadIdx.x + st];
    }
    __syncthreads();
  }
  if (threadIdx.x == 0) {
    float mu = sh[0] / (float)n;
    float var = sh2[0] / (float)n - mu * mu;
    stats[bg * 2] = mu;
    stats[bg * 2 + 1] = var;
  }
}

__global__ void __launch_bounds__(256) gn_apply_kernel(const float* __restrict__ merged,
                                                       const float* __restrict__ stats,
                                                       const float* __restrict__ gw,
                                                       const float* __restrict__ gb,
                                                       float* __restrict__ out) {
  int i = blockIdx.x * blockDim.x + threadIdx.x;
  if (i >= BATCH * CDIM * HWN) return;
  int c = (i / HWN) % CDIM;
  int b = i / (HWN * CDIM);
  int g = c / (CDIM / GN_GROUPS);
  float mu = stats[(b * GN_GROUPS + g) * 2];
  float var = stats[(b * GN_GROUPS + g) * 2 + 1];
  float v = (merged[i] - mu) * rsqrtf(var + GN_EPS) * gw[c] + gb[c];
  out[i] = fmaxf(v, 0.f);
}

// ---------------- launch ----------------
extern "C" void kernel_launch(void* const* d_in, const int* in_sizes, int n_in,
                              void* d_out, int out_size, void* d_ws, size_t ws_size,
                              hipStream_t stream) {
  (void)in_sizes; (void)n_in; (void)out_size; (void)ws_size;
  const float* x = (const float*)d_in[0];
  const float* offs = (const float*)d_in[1];
  const float* ef_w = (const float*)d_in[2];
  const float* ef_b = (const float*)d_in[3];
  const float* hm1_w = (const float*)d_in[4];
  const float* hm1_b = (const float*)d_in[5];
  const float* hm2_w = (const float*)d_in[6];
  const float* hm2_b = (const float*)d_in[7];
  const float* mrg_w = (const float*)d_in[8];
  const float* mrg_b = (const float*)d_in[9];
  const float* gn_g = (const float*)d_in[10];
  const float* gn_b = (const float*)d_in[11];
  float* out = (float*)d_out;

  char* ws = (char*)d_ws;
  size_t off = 0;
  auto take = [&](size_t bytes) -> char* {
    char* p = ws + off;
    off = (off + bytes + 255) & ~(size_t)255;
    return p;
  };
  __bf16* xt = (__bf16*)take((size_t)BATCH * HWN * CDIM * 2);     // x transposed bf16
  __bf16* efw = (__bf16*)take((size_t)DDIM * CDIM * 2);           // ef_w bf16
  __bf16* h1w = (__bf16*)take((size_t)PNUM * CDIM * CDIM * 2);    // hm1_w bf16
  __bf16* mww = (__bf16*)take((size_t)CDIM * DDIM * 2);           // merge_w bf16
  __bf16* eft = (__bf16*)take((size_t)BATCH * HWN * DDIM * 2);    // ef, N-major bf16
  float* heat = (float*)take((size_t)BATCH * PNUM * HWN * 4);     // heat acc / heat
  __bf16* st = (__bf16*)take((size_t)BATCH * HWN * DDIM * 2);     // sampled outs N-major
  float* merged = (float*)take((size_t)BATCH * CDIM * HWN * 4);   // merge output fp32
  float* stats = (float*)take((size_t)BATCH * GN_GROUPS * 2 * 4); // GN mean/var

  cvt_kernel<<<(DDIM * CDIM + 255) / 256, 256, 0, stream>>>(ef_w, efw, DDIM * CDIM);
  cvt_kernel<<<(PNUM * CDIM * CDIM + 255) / 256, 256, 0, stream>>>(hm1_w, h1w,
                                                                   PNUM * CDIM * CDIM);
  cvt_kernel<<<(CDIM * DDIM + 255) / 256, 256, 0, stream>>>(mrg_w, mww, CDIM * DDIM);
  cvt_x_kernel<<<(BATCH * HWN * CDIM + 255) / 256, 256, 0, stream>>>(x, xt);
  zero_kernel<<<(BATCH * PNUM * HWN + 255) / 256, 256, 0, stream>>>(
      heat, BATCH * PNUM * HWN);

  gemm_ef_kernel<<<(BATCH * (DDIM / 16) * NTILES) / 8, 256, 0, stream>>>(efw, xt,
                                                                         ef_b, eft);
  gemm_heat_kernel<<<(BATCH * PNUM * (CDIM / 16) * NTILES) / 8, 256, 0, stream>>>(
      h1w, eft, hm1_b, hm2_w, heat);
  heat_exp_kernel<<<(BATCH * PNUM * HWN + 255) / 256, 256, 0, stream>>>(heat, hm2_b);
  sample_kernel<<<BATCH * PNUM * HWN, 256, 0, stream>>>(eft, heat, offs, st);
  gemm_merge_kernel<<<(BATCH * (CDIM / 16) * NTILES) / 8, 256, 0, stream>>>(
      mww, st, mrg_b, merged);
  gn_reduce_kernel<<<BATCH * GN_GROUPS, 256, 0, stream>>>(merged, stats);
  gn_apply_kernel<<<(BATCH * CDIM * HWN + 255) / 256, 256, 0, stream>>>(
      merged, stats, gn_g, gn_b, out);
}
